// GraphUNet_84817014161898
// MI455X (gfx1250) — matrix operand order repaired and checked by reading, not compiled, not run on hardware
//
#include <hip/hip_runtime.h>
#include <hip/hip_bf16.h>

typedef __attribute__((ext_vector_type(16))) __bf16 v16bf;
typedef __attribute__((ext_vector_type(8)))  float  v8f;

#define H 256          // hidden width (all GEMM outputs are 256 wide)
#define WAVE 32

__device__ __forceinline__ unsigned short f2bfbits(float f) {
    unsigned u = __builtin_bit_cast(unsigned, f);
    unsigned r = u + 0x7FFFu + ((u >> 16) & 1u);   // round-to-nearest-even
    return (unsigned short)(r >> 16);
}
__device__ __forceinline__ __bf16 f2bf(float f) {
    unsigned short h = f2bfbits(f);
    return __builtin_bit_cast(__bf16, h);
}

// ---------------------------------------------------------------------------
// Pack W[cin,256] (f32, row-major) into bf16 WMMA B-fragment order:
//   Wp[((kb*2 + half)*256 + col)*16 + e] = bf16( W[(kb*32 + half*16 + e)*256 + col] )
// so a lane's whole B fragment is one contiguous 32-byte load.
// ---------------------------------------------------------------------------
__global__ void pack_w_bf16(const float* __restrict__ W,
                            unsigned short* __restrict__ Wp, int cin) {
    const int total = cin * H;                       // packed element count
    for (int i = blockIdx.x * blockDim.x + threadIdx.x; i < total;
         i += gridDim.x * blockDim.x) {
        const int e   = i & 15;
        const int col = (i >> 4) & 255;
        const int h   = (i >> 12) & 1;
        const int kb  = i >> 13;
        const int K   = kb * 32 + h * 16 + e;
        Wp[i] = f2bfbits(W[(size_t)K * H + col]);
    }
}

// ---------------------------------------------------------------------------
// GEMM: out[n,256] = concat(A1,A2)[n,cin] @ W[cin,256] + bias
// bf16 WMMA, f32 accumulate. Wave computes a 16x32 strip (2 accumulators,
// shared A fragment); block = 8 waves = 128x32 tile. K-loop unrolled x2.
// ---------------------------------------------------------------------------
__global__ void gemm_bias_wmma(const float* __restrict__ A1,
                               const float* __restrict__ A2,
                               int cin1, int cin,
                               const unsigned short* __restrict__ Wp,
                               const float* __restrict__ bias,
                               float* __restrict__ out, int n) {
    const int lane = threadIdx.x & 31;
    const int wave = threadIdx.x >> 5;
    const int rowBase = blockIdx.x * 128 + wave * 16;
    const int colBase = blockIdx.y * 32;
    if (rowBase >= n) return;                 // wave-uniform exit, EXEC stays full

    int row = rowBase + (lane & 15);
    if (row >= n) row = n - 1;                // clamp loads; stores are guarded
    const int half  = lane >> 4;
    const int kgrpA = half * 16;              // A: K offset group (contiguous, matches packed B)
    const int col0  = colBase + (lane & 15);
    const int cin2  = cin - cin1;

    v8f acc0 = {}, acc1 = {};
    const int nkb = cin >> 5;
#pragma unroll 2
    for (int kb = 0; kb < nkb; ++kb) {
        const int k0 = kb * 32;
        const float* Ap; int kl;
        if (k0 < cin1) { Ap = A1 + (long)row * cin1; kl = k0; }
        else           { Ap = A2 + (long)row * cin2; kl = k0 - cin1; }
        v16bf a;
#pragma unroll
        for (int i = 0; i < 8; ++i) {
            a[i]     = f2bf(Ap[kl + kgrpA + i]);        // K = kbase + i
            a[i + 8] = f2bf(Ap[kl + kgrpA + 8 + i]);    // K = kbase + 8 + i
        }
        // B fragments: contiguous 32B per lane from packed layout
        const v16bf* bp = (const v16bf*)(Wp + ((size_t)(kb * 2 + half) * H) * 16);
        const v16bf b0 = bp[col0];
        const v16bf b1 = bp[col0 + 16];
        acc0 = __builtin_amdgcn_wmma_f32_16x16x32_bf16(false, a, false, b0,
                                                       (short)0, acc0, false, false);
        acc1 = __builtin_amdgcn_wmma_f32_16x16x32_bf16(false, a, false, b1,
                                                       (short)0, acc1, false, false);
    }
    const float bc0 = bias ? bias[col0] : 0.f;
    const float bc1 = bias ? bias[col0 + 16] : 0.f;
    if (rowBase + 16 <= n) {                  // full tile: unguarded stores
#pragma unroll
        for (int r = 0; r < 8; ++r) {         // D: M = r + 8*half
            const long orow = rowBase + r + half * 8;
            out[orow * H + col0]      = acc0[r] + bc0;
            out[orow * H + col0 + 16] = acc1[r] + bc1;
        }
    } else {                                  // tail tile: guarded
#pragma unroll
        for (int r = 0; r < 8; ++r) {
            const int orow = rowBase + r + half * 8;
            if (orow < n) {
                out[(long)orow * H + col0]      = acc0[r] + bc0;
                out[(long)orow * H + col0 + 16] = acc1[r] + bc1;
            }
        }
    }
}

// ---------------------------------------------------------------------------
// Edge scatter: tot[dst] += ew * m[src]; cnt[dst] += 1   (32 lanes / edge)
// ---------------------------------------------------------------------------
__global__ void sage_aggregate(const float* __restrict__ m,
                               const int* __restrict__ src,
                               const int* __restrict__ dst,
                               const float* __restrict__ ew,
                               const unsigned char* __restrict__ valid,
                               float* __restrict__ tot, float* __restrict__ cnt,
                               int E) {
    const int lane   = threadIdx.x & 31;
    const int stride = gridDim.x * (blockDim.x >> 5);
    int e = blockIdx.x * (blockDim.x >> 5) + (threadIdx.x >> 5);
    for (; e < E; e += stride) {
        int en = e + stride;                          // prefetch next source row
        if (en < E) __builtin_prefetch(m + (long)src[en] * H + lane * 8, 0, 1);
        if (valid && !valid[e]) continue;
        const int s = src[e], d = dst[e];
        const float w = ew[e];
        const float* ms = m + (long)s * H + lane * 8;
        float* td = tot + (long)d * H + lane * 8;
#pragma unroll
        for (int i = 0; i < 8; ++i) atomicAdd(td + i, ms[i] * w);
        if (lane == 0) atomicAdd(&cnt[d], 1.0f);
    }
}

// ---------------------------------------------------------------------------
// Finalize: mean (+ self loop) [+ identity] [-> LayerNorm -> GELU]
// one wave per row, 8 features per lane
// ---------------------------------------------------------------------------
__global__ void sage_post(const float* __restrict__ tot, const float* __restrict__ cnt,
                          const float* __restrict__ m, const float* __restrict__ ident,
                          const float* __restrict__ g, const float* __restrict__ beta,
                          float* __restrict__ out, int n, int selfloop, int lngelu) {
    const int lane   = threadIdx.x & 31;
    const int stride = gridDim.x * (blockDim.x >> 5);
    int row = blockIdx.x * (blockDim.x >> 5) + (threadIdx.x >> 5);
    for (; row < n; row += stride) {
        const long base = (long)row * H + lane * 8;
        const float inv = 1.f / fmaxf(cnt[row] + (selfloop ? 1.f : 0.f), 1.f);
        float v[8], s = 0.f;
#pragma unroll
        for (int i = 0; i < 8; ++i) {
            float h = tot[base + i];
            if (selfloop) h += m[base + i];
            h *= inv;
            if (ident) h += ident[base + i];
            v[i] = h; s += h;
        }
        if (lngelu) {
            for (int o = 16; o; o >>= 1) s += __shfl_xor(s, o, WAVE);
            const float mu = s * (1.f / 256.f);
            float var = 0.f;
#pragma unroll
            for (int i = 0; i < 8; ++i) { float d = v[i] - mu; var += d * d; }
            for (int o = 16; o; o >>= 1) var += __shfl_xor(var, o, WAVE);
            const float r = rsqrtf(var * (1.f / 256.f) + 1e-5f);
#pragma unroll
            for (int i = 0; i < 8; ++i) {
                const int j = lane * 8 + i;
                const float y = (v[i] - mu) * r * g[j] + beta[j];
                out[base + i] = 0.5f * y * (1.f + erff(y * 0.70710678118654752f));
            }
        } else {
#pragma unroll
            for (int i = 0; i < 8; ++i) out[base + i] = v[i];
        }
    }
}

// ---------------------------------------------------------------------------
// TopK pooling pieces
// ---------------------------------------------------------------------------
__global__ void wnorm_kernel(const float* __restrict__ w, float* __restrict__ inv) {
    __shared__ float s[256];
    float v = w[threadIdx.x];
    s[threadIdx.x] = v * v; __syncthreads();
    for (int o = 128; o; o >>= 1) {
        if ((int)threadIdx.x < o) s[threadIdx.x] += s[threadIdx.x + o];
        __syncthreads();
    }
    if (threadIdx.x == 0) inv[0] = rsqrtf(s[0]);
}

__global__ void pool_score(const float* __restrict__ x, const float* __restrict__ w,
                           const float* __restrict__ invn, float* __restrict__ score,
                           unsigned* __restrict__ keys, int n) {
    const int lane   = threadIdx.x & 31;
    const int stride = gridDim.x * (blockDim.x >> 5);
    int row = blockIdx.x * (blockDim.x >> 5) + (threadIdx.x >> 5);
    for (; row < n; row += stride) {
        const long base = (long)row * H + lane * 8;
        float s = 0.f;
#pragma unroll
        for (int i = 0; i < 8; ++i) s += x[base + i] * w[lane * 8 + i];
        for (int o = 16; o; o >>= 1) s += __shfl_xor(s, o, WAVE);
        if (lane == 0) {
            const float sc = tanhf(s * invn[0]);
            score[row] = sc;
            unsigned u = __builtin_bit_cast(unsigned, sc);
            keys[row] = (u & 0x80000000u) ? ~u : (u | 0x80000000u); // order-preserving
        }
    }
}

// sel: [0]=prefix [1]=remaining [2]=posCounter [3]=tieCounter [4]=thresholdKey
__global__ void topk_init(unsigned* __restrict__ sel, unsigned* __restrict__ hist,
                          unsigned k) {
    hist[threadIdx.x] = 0u;
    if (threadIdx.x < 8) sel[threadIdx.x] = 0u;
    if (threadIdx.x == 1) sel[1] = k;
}

__global__ void topk_hist(const unsigned* __restrict__ keys, const unsigned* __restrict__ sel,
                          unsigned* __restrict__ hist, int n, int shift) {
    const unsigned prefix = sel[0];
    const unsigned himask = (shift == 24) ? 0u : (0xFFFFFFFFu << (shift + 8));
    for (int i = blockIdx.x * blockDim.x + threadIdx.x; i < n;
         i += gridDim.x * blockDim.x) {
        const unsigned key = keys[i];
        if ((key & himask) == (prefix & himask))
            atomicAdd(&hist[(key >> shift) & 255u], 1u);
    }
}

__global__ void topk_scan(unsigned* __restrict__ sel, unsigned* __restrict__ hist,
                          int shift) {
    __shared__ unsigned h[256];
    h[threadIdx.x] = hist[threadIdx.x];
    __syncthreads();
    if (threadIdx.x == 0) {
        const unsigned remaining = sel[1];
        unsigned cum = 0; int b = 255;
        for (; b >= 0; --b) { cum += h[b]; if (cum >= remaining) break; }
        if (b < 0) b = 0;
        const unsigned above = cum - h[b];
        sel[0] |= ((unsigned)b) << shift;
        sel[1] = remaining - above;
        if (shift == 0) sel[4] = sel[0];
    }
    hist[threadIdx.x] = 0u;                      // reset for next pass
}

__global__ void topk_compact(const unsigned* __restrict__ keys, unsigned* __restrict__ sel,
                             int* __restrict__ perm, int* __restrict__ mapping,
                             int n, int k) {
    const unsigned T = sel[4], rem = sel[1];
    for (int i = blockIdx.x * blockDim.x + threadIdx.x; i < n;
         i += gridDim.x * blockDim.x) {
        const unsigned key = keys[i];
        int pos = -1;
        if (key > T) pos = (int)atomicAdd(&sel[2], 1u);
        else if (key == T && atomicAdd(&sel[3], 1u) < rem)
            pos = (int)atomicAdd(&sel[2], 1u);
        if (pos >= 0 && pos < k) { perm[pos] = i; mapping[i] = pos; }
    }
}

__global__ void pool_gather(const float* __restrict__ xin, const float* __restrict__ score,
                            const int* __restrict__ perm, float* __restrict__ xout, int k) {
    const int lane   = threadIdx.x & 31;
    const int stride = gridDim.x * (blockDim.x >> 5);
    int p = blockIdx.x * (blockDim.x >> 5) + (threadIdx.x >> 5);
    for (; p < k; p += stride) {
        const int srcrow = perm[p];
        const float v = score[srcrow];
        const float* si = xin + (long)srcrow * H + lane * 8;
        float* so = xout + (long)p * H + lane * 8;
#pragma unroll
        for (int i = 0; i < 8; ++i) so[i] = si[i] * v;
    }
}

__global__ void relabel_edges(const int* __restrict__ si, const int* __restrict__ di,
                              const unsigned char* __restrict__ vi,
                              const int* __restrict__ mapping,
                              int* __restrict__ so, int* __restrict__ dp,
                              unsigned char* __restrict__ vo, int E) {
    for (int e = blockIdx.x * blockDim.x + threadIdx.x; e < E;
         e += gridDim.x * blockDim.x) {
        int ok = vi ? (int)vi[e] : 1;
        const int s = mapping[si[e]], d = mapping[di[e]];
        ok = ok && (s >= 0) && (d >= 0);
        so[e] = ok ? s : 0; dp[e] = ok ? d : 0; vo[e] = (unsigned char)ok;
    }
}

__global__ void unpool_scatter(const float* __restrict__ xs, const int* __restrict__ perm,
                               float* __restrict__ up, int k) {
    const int lane   = threadIdx.x & 31;
    const int stride = gridDim.x * (blockDim.x >> 5);
    int p = blockIdx.x * (blockDim.x >> 5) + (threadIdx.x >> 5);
    for (; p < k; p += stride) {
        const int d = perm[p];
        const float* si = xs + (long)p * H + lane * 8;
        float* so = up + (long)d * H + lane * 8;
#pragma unroll
        for (int i = 0; i < 8; ++i) so[i] = si[i];
    }
}

__global__ void global_pool(const float* __restrict__ x, float* __restrict__ out, int n) {
    const int j = blockIdx.x;                       // feature column
    float s = 0.f, mx = -3.0e38f;
    for (int i = threadIdx.x; i < n; i += blockDim.x) {
        const float v = x[(long)i * H + j];
        s += v; mx = fmaxf(mx, v);
    }
    __shared__ float ss[256], sm[256];
    ss[threadIdx.x] = s; sm[threadIdx.x] = mx; __syncthreads();
    for (int o = 128; o; o >>= 1) {
        if ((int)threadIdx.x < o) {
            ss[threadIdx.x] += ss[threadIdx.x + o];
            sm[threadIdx.x] = fmaxf(sm[threadIdx.x], sm[threadIdx.x + o]);
        }
        __syncthreads();
    }
    if (threadIdx.x == 0) { out[j] = ss[0] / (float)n; out[H + j] = sm[0]; }
}

__global__ void fill_f32(float* __restrict__ p, float v, size_t cnt) {
    for (size_t i = (size_t)blockIdx.x * blockDim.x + threadIdx.x; i < cnt;
         i += (size_t)gridDim.x * blockDim.x) p[i] = v;
}
__global__ void fill_i32(int* __restrict__ p, int v, size_t cnt) {
    for (size_t i = (size_t)blockIdx.x * blockDim.x + threadIdx.x; i < cnt;
         i += (size_t)gridDim.x * blockDim.x) p[i] = v;
}

// ---------------------------------------------------------------------------
extern "C" void kernel_launch(void* const* d_in, const int* in_sizes, int n_in,
                              void* d_out, int out_size, void* d_ws, size_t ws_size,
                              hipStream_t stream) {
    const int N = 50000, E = 800000, CIN = 1280;
    const int n1 = 25000, n2 = 12500, n3 = 6250;

    const float* x_in = (const float*)d_in[0];
    const int*   ei   = (const int*)d_in[1];
    const float* ew   = (const float*)d_in[2];
    const int* src0 = ei, * dst0 = ei + E;

    auto P = [&](int i) { return (const float*)d_in[i]; };
    // params flat order: enc0_W,enc0_b, enc0_res{W,b,g,beta}, pool0..2_w,
    // enc1..enc3{W,b,g,beta}, bn0,bn1{..}, dec0..dec2{W,b,g,beta,rW,rb}
    const float* enc0_W = P(3), * enc0_b = P(4);
    const int R_ENC0 = 5, POOLW = 9, R_ENC1 = 12, R_ENC2 = 16, R_ENC3 = 20,
              R_BN0 = 24, R_BN1 = 28, R_DEC0 = 32, R_DEC1 = 38, R_DEC2 = 44;

    // --- workspace carve (deterministic) ---
    char* wp = (char*)d_ws;
    auto carve = [&](size_t bytes) -> void* {
        void* p = wp; wp += (bytes + 255) & ~(size_t)255; return p;
    };
    float* m     = (float*)carve((size_t)N * H * 4);
    float* tot   = (float*)carve((size_t)N * H * 4);
    float* xA    = (float*)carve((size_t)N * H * 4);
    float* xB    = (float*)carve((size_t)N * H * 4);
    float* ident = (float*)carve((size_t)N * H * 4);
    float* up    = (float*)carve((size_t)N * H * 4);
    float* xs0   = (float*)carve((size_t)N * H * 4);
    float* xs1   = (float*)carve((size_t)n1 * H * 4);
    float* xs2   = (float*)carve((size_t)n2 * H * 4);
    float* cnt   = (float*)carve((size_t)N * 4);
    float* score = (float*)carve((size_t)N * 4);
    unsigned* keys = (unsigned*)carve((size_t)N * 4);
    int* perm0 = (int*)carve((size_t)n1 * 4);
    int* perm1 = (int*)carve((size_t)n2 * 4);
    int* perm2 = (int*)carve((size_t)n3 * 4);
    int* mapping = (int*)carve((size_t)N * 4);
    int* src1 = (int*)carve((size_t)E * 4); int* dst1 = (int*)carve((size_t)E * 4);
    int* src2 = (int*)carve((size_t)E * 4); int* dst2 = (int*)carve((size_t)E * 4);
    int* src3 = (int*)carve((size_t)E * 4); int* dst3 = (int*)carve((size_t)E * 4);
    unsigned char* val1 = (unsigned char*)carve(E);
    unsigned char* val2 = (unsigned char*)carve(E);
    unsigned char* val3 = (unsigned char*)carve(E);
    unsigned* hist = (unsigned*)carve(256 * 4);
    unsigned* sel  = (unsigned*)carve(8 * 4);
    float* invn    = (float*)carve(4);
    unsigned short* Wp = (unsigned short*)carve((size_t)CIN * H * 2); // packed B

    // --- helpers ---
    auto fillf = [&](float* p, float v, size_t c) {
        fill_f32<<<1024, 256, 0, stream>>>(p, v, c);
    };
    auto gemm = [&](const float* A1, const float* A2, int cin1, int cin,
                    const float* W, const float* b, float* o, int n) {
        pack_w_bf16<<<512, 256, 0, stream>>>(W, Wp, cin);
        dim3 g((n + 127) / 128, H / 32);
        gemm_bias_wmma<<<g, 256, 0, stream>>>(A1, A2, cin1, cin, Wp, b, o, n);
    };
    auto aggregate = [&](const float* mm, const int* s, const int* d,
                         const unsigned char* v, int n) {
        fillf(tot, 0.f, (size_t)n * H);
        fillf(cnt, 0.f, (size_t)n);
        sage_aggregate<<<2048, 256, 0, stream>>>(mm, s, d, ew, v, tot, cnt, E);
    };
    auto resblock = [&](const float* A1, const float* A2, int cin1, int cin,
                        int pidx, const float* id,
                        const int* s, const int* d, const unsigned char* v,
                        float* o, int n) {
        gemm(A1, A2, cin1, cin, P(pidx), P(pidx + 1), m, n);
        aggregate(m, s, d, v, n);
        sage_post<<<1024, 256, 0, stream>>>(tot, cnt, m, id, P(pidx + 2),
                                            P(pidx + 3), o, n, 1, 1);
    };
    auto pool = [&](const float* xin, const float* wv, int n, int k, int* perm,
                    const int* si, const int* di, const unsigned char* vi,
                    int* so, int* dp, unsigned char* vo, float* xout) {
        wnorm_kernel<<<1, 256, 0, stream>>>(wv, invn);
        pool_score<<<1024, 256, 0, stream>>>(xin, wv, invn, score, keys, n);
        topk_init<<<1, 256, 0, stream>>>(sel, hist, (unsigned)k);
        for (int pass = 0; pass < 4; ++pass) {
            const int shift = 24 - 8 * pass;
            topk_hist<<<1024, 256, 0, stream>>>(keys, sel, hist, n, shift);
            topk_scan<<<1, 256, 0, stream>>>(sel, hist, shift);
        }
        fill_i32<<<1024, 256, 0, stream>>>(mapping, -1, (size_t)n);
        topk_compact<<<1024, 256, 0, stream>>>(keys, sel, perm, mapping, n, k);
        pool_gather<<<1024, 256, 0, stream>>>(xin, score, perm, xout, k);
        relabel_edges<<<2048, 256, 0, stream>>>(si, di, vi, mapping, so, dp, vo, E);
    };

    // --- forward ---
    // enc0: plain weighted-SAGE mean (no self loops, no LN/GELU)
    gemm(x_in, nullptr, CIN, CIN, enc0_W, enc0_b, m, N);
    aggregate(m, src0, dst0, nullptr, N);
    sage_post<<<1024, 256, 0, stream>>>(tot, cnt, m, nullptr, nullptr, nullptr,
                                        xA, N, 0, 0);
    // enc0_res (self loops + LN + GELU), identity = x
    resblock(xA, nullptr, H, H, R_ENC0, xA, src0, dst0, nullptr, xB, N);
    hipMemcpyAsync(xs0, xB, (size_t)N * H * 4, hipMemcpyDeviceToDevice, stream);

    // pool0 -> enc1
    pool(xB, P(POOLW + 0), N, n1, perm0, src0, dst0, nullptr, src1, dst1, val1, xA);
    resblock(xA, nullptr, H, H, R_ENC1, xA, src1, dst1, val1, xB, n1);
    hipMemcpyAsync(xs1, xB, (size_t)n1 * H * 4, hipMemcpyDeviceToDevice, stream);

    // pool1 -> enc2
    pool(xB, P(POOLW + 1), n1, n2, perm1, src1, dst1, val1, src2, dst2, val2, xA);
    resblock(xA, nullptr, H, H, R_ENC2, xA, src2, dst2, val2, xB, n2);
    hipMemcpyAsync(xs2, xB, (size_t)n2 * H * 4, hipMemcpyDeviceToDevice, stream);

    // pool2 -> enc3
    pool(xB, P(POOLW + 2), n2, n3, perm2, src2, dst2, val2, src3, dst3, val3, xA);
    resblock(xA, nullptr, H, H, R_ENC3, xA, src3, dst3, val3, xB, n3);

    // bottleneck
    resblock(xB, nullptr, H, H, R_BN0, xB, src3, dst3, val3, xA, n3);
    resblock(xA, nullptr, H, H, R_BN1, xA, src3, dst3, val3, xB, n3);

    // decode i=0 (level j=2, n=12500)
    fillf(up, 0.f, (size_t)n2 * H);
    unpool_scatter<<<1024, 256, 0, stream>>>(xB, perm2, up, n3);
    gemm(xs2, up, H, 2 * H, P(R_DEC0 + 4), P(R_DEC0 + 5), ident, n2);   // rW,rb
    resblock(xs2, up, H, 2 * H, R_DEC0, ident, src2, dst2, val2, xA, n2);

    // decode i=1 (level j=1, n=25000)
    fillf(up, 0.f, (size_t)n1 * H);
    unpool_scatter<<<1024, 256, 0, stream>>>(xA, perm1, up, n2);
    gemm(xs1, up, H, 2 * H, P(R_DEC1 + 4), P(R_DEC1 + 5), ident, n1);
    resblock(xs1, up, H, 2 * H, R_DEC1, ident, src1, dst1, val1, xB, n1);

    // decode i=2 (level j=0, n=50000)
    fillf(up, 0.f, (size_t)N * H);
    unpool_scatter<<<1024, 256, 0, stream>>>(xB, perm0, up, n1);
    gemm(xs0, up, H, 2 * H, P(R_DEC2 + 4), P(R_DEC2 + 5), ident, N);
    resblock(xs0, up, H, 2 * H, R_DEC2, ident, src0, dst0, nullptr, xA, N);

    // global mean/max pool -> [1, 512]
    global_pool<<<256, 256, 0, stream>>>(xA, (float*)d_out, N);
}
